// Intermodule_84739704750828
// MI455X (gfx1250) — compile-verified, matrix operand order checked by reading, not run on hardware
//
#include <hip/hip_runtime.h>
#include <hip/hip_bf16.h>

#define BB 4
#define CC 128
#define NN 4096
#define JT 64            // j-tile width per iteration
#define NJT (NN / JT)    // 64 iterations
#define KLDE 136         // padded LDS row stride (elements) for K tile rows (128 + 8)
#define VLDE 72          // padded LDS row stride for V tile rows (64 + 8)

// CDNA5 async global->LDS staging (ASYNCcnt path); fall back to plain
// load+ds_store if this toolchain doesn't expose the builtins.
#if defined(__AMDGCN__) && __has_builtin(__builtin_amdgcn_global_load_async_to_lds_b128) && \
    __has_builtin(__builtin_amdgcn_s_wait_asynccnt)
#define USE_ASYNC 1
#else
#define USE_ASYNC 0
#endif

typedef __attribute__((ext_vector_type(16))) __bf16 v16bf;
typedef __attribute__((ext_vector_type(8)))  float  v8f;
typedef __attribute__((ext_vector_type(4)))  int    v4i;

#define LOG2E 1.44269504088896340736f

union FragU { v16bf bf; uint4 u[2]; };

// Load a 16x32 bf16 A-style fragment (also B-style: lane = column) from a
// row-major matrix with leading dimension `ld` (elements).
// CDNA5 ISA 7.12.2: lanes 0-15 hold K={0..7,16..23}, lanes 16-31 hold
// K={8..15,24..31}; row/col = lane&15.  Two aligned b128 loads per lane.
__device__ __forceinline__ v16bf load_frag(const __bf16* base, int ld) {
    const int lane = threadIdx.x & 31;
    const int r    = lane & 15;
    const int kb   = (lane & 16) ? 8 : 0;
    const __bf16* p = base + (size_t)r * ld + kb;
    FragU f;
    f.u[0] = *reinterpret_cast<const uint4*>(p);
    f.u[1] = *reinterpret_cast<const uint4*>(p + 16);
    return f.bf;
}

// Stage one K tile (64 rows x 128 ch, contiguous 16KB) and one V tile
// (128 ch-rows x 64 j, row stride NN) into padded LDS. 16 b128 chunks/thread.
__device__ __forceinline__ void stage_kv(const __bf16* Kg, const __bf16* Vg,
                                         __bf16* Kl, __bf16* Vl, int t)
{
#pragma unroll
    for (int c = 0; c < 8; ++c) {
        const int chunk = c * 128 + t;
        const int krow = chunk >> 4, kw = chunk & 15;   // 64 rows x 16 chunks
        const int vrow = chunk >> 3, vw = chunk & 7;    // 128 rows x 8 chunks
#if USE_ASYNC
        __builtin_amdgcn_global_load_async_to_lds_b128(
            (__attribute__((address_space(1))) v4i*)((char*)Kg + krow * 256 + kw * 16),
            (__attribute__((address_space(3))) v4i*)((char*)Kl + krow * (KLDE * 2) + kw * 16),
            0, 0);
        __builtin_amdgcn_global_load_async_to_lds_b128(
            (__attribute__((address_space(1))) v4i*)((char*)Vg + (size_t)vrow * (NN * 2) + vw * 16),
            (__attribute__((address_space(3))) v4i*)((char*)Vl + vrow * (VLDE * 2) + vw * 16),
            0, 0);
#else
        *(uint4*)((char*)Kl + krow * (KLDE * 2) + kw * 16) =
            *(const uint4*)((const char*)Kg + krow * 256 + kw * 16);
        *(uint4*)((char*)Vl + vrow * (VLDE * 2) + vw * 16) =
            *(const uint4*)((const char*)Vg + (size_t)vrow * (NN * 2) + vw * 16);
#endif
    }
}

// ---------------------------------------------------------------------------
// Kernel 1: 1x1-conv QKV projection (memory-bound, ~4% of FLOPs).
// Q,K stored [B,N,C] bf16 (fragment rows contiguous); V stored [B,C,N] bf16.
// Q is pre-scaled by log2(e) so the attention softmax runs in base 2 with a
// single native v_exp_f32 per score.
// ---------------------------------------------------------------------------
__global__ __launch_bounds__(128) void qkv_proj(
    const float* __restrict__ x,
    const float* __restrict__ Wq, const float* __restrict__ bq,
    const float* __restrict__ Wk, const float* __restrict__ bk,
    const float* __restrict__ Wv, const float* __restrict__ bv,
    __bf16* __restrict__ qm, __bf16* __restrict__ km, __bf16* __restrict__ vm)
{
    __shared__ float xt[CC][16];
    const int t  = threadIdx.x;            // output channel
    const int b  = blockIdx.x >> 8;        // 256 position-tiles per batch
    const int i0 = (blockIdx.x & 255) << 4;

    const float4* xr4 = reinterpret_cast<const float4*>(
        x + ((size_t)b * CC + t) * NN + i0);
    float4 v0 = xr4[0], v1 = xr4[1], v2 = xr4[2], v3 = xr4[3];
    xt[t][ 0] = v0.x; xt[t][ 1] = v0.y; xt[t][ 2] = v0.z; xt[t][ 3] = v0.w;
    xt[t][ 4] = v1.x; xt[t][ 5] = v1.y; xt[t][ 6] = v1.z; xt[t][ 7] = v1.w;
    xt[t][ 8] = v2.x; xt[t][ 9] = v2.y; xt[t][10] = v2.z; xt[t][11] = v2.w;
    xt[t][12] = v3.x; xt[t][13] = v3.y; xt[t][14] = v3.z; xt[t][15] = v3.w;
    __syncthreads();

    float aq[16], ak[16], av[16];
    const float bqv = bq[t], bkv = bk[t], bvv = bv[t];
#pragma unroll
    for (int ii = 0; ii < 16; ++ii) { aq[ii] = bqv; ak[ii] = bkv; av[ii] = bvv; }

    for (int k = 0; k < CC; ++k) {
        const float wq = Wq[t * CC + k];
        const float wk = Wk[t * CC + k];
        const float wv = Wv[t * CC + k];
#pragma unroll
        for (int ii = 0; ii < 16; ++ii) {
            const float xv = xt[k][ii];
            aq[ii] = fmaf(wq, xv, aq[ii]);
            ak[ii] = fmaf(wk, xv, ak[ii]);
            av[ii] = fmaf(wv, xv, av[ii]);
        }
    }

#pragma unroll
    for (int ii = 0; ii < 16; ++ii) {
        const size_t row = (size_t)b * NN + i0 + ii;
        qm[row * CC + t] = (__bf16)(aq[ii] * LOG2E);   // base-2 softmax domain
        km[row * CC + t] = (__bf16)ak[ii];
        vm[((size_t)b * CC + t) * NN + i0 + ii] = (__bf16)av[ii];
    }
}

// ---------------------------------------------------------------------------
// Kernel 2: flash attention with transposed score tiles (S^T = K Q^T).
// Lane L holds all scores for query row i=L (split with partner lane L+16),
// so the online softmax is lane-local (one shfl_xor(16) per tile) and the
// S^T accumulator IS the A-layout P fragment (no LDS transpose bounce).
// Block = 4 waves; K/V j-tiles (64 wide) double-buffered in LDS via async
// global->LDS copies. 32 bf16 WMMAs per iteration.
// ---------------------------------------------------------------------------
__global__ __launch_bounds__(128) void attn_fa(
    const __bf16* __restrict__ qm, const __bf16* __restrict__ km,
    const __bf16* __restrict__ vm, const float* __restrict__ x,
    float* __restrict__ out)
{
    __shared__ __bf16 Kt[2][JT * KLDE];     // 2 x 17KB
    __shared__ __bf16 Vt[2][CC * VLDE];     // 2 x 18KB

    const int t    = threadIdx.x;
    const int lane = t & 31;
    const int wv   = t >> 5;
    const int tile = blockIdx.x * 4 + wv;
    const int b    = tile >> 8;             // 256 tiles of 16 rows per batch
    const int i0   = (tile & 255) << 4;

    const int colc  = lane & 15;
    const int rhalf = (lane & 16) ? 8 : 0;

    // Resident Q fragments: 16 rows x 128 K (4 fragments of K=32).
    const __bf16* Qb = qm + ((size_t)b * NN + i0) * CC;
    v16bf qf[4];
#pragma unroll
    for (int kc = 0; kc < 4; ++kc) qf[kc] = load_frag(Qb + kc * 32, CC);

    v8f O[8];
#pragma unroll
    for (int ct = 0; ct < 8; ++ct)
#pragma unroll
        for (int e = 0; e < 8; ++e) O[ct][e] = 0.0f;

    float mlane = -3.0e38f;                 // running max (log2 domain)
    float llane = 0.0f;                     // per-lane partial row sum

    const __bf16* Kbase = km + (size_t)b * NN * CC;
    const __bf16* Vbase = vm + (size_t)b * CC * NN;

    // Prologue: stage first j-tile into buffer 0.
    stage_kv(Kbase, Vbase, &Kt[0][0], &Vt[0][0], t);

    for (int jt = 0; jt < NJT; ++jt) {
        const int cur = jt & 1;
        if (jt + 1 < NJT) {
            stage_kv(Kbase + (size_t)(jt + 1) * JT * CC, Vbase + (jt + 1) * JT,
                     &Kt[cur ^ 1][0], &Vt[cur ^ 1][0], t);
#if USE_ASYNC
            // 16 new async ops in flight for the next buffer; previous tile's
            // 16 are guaranteed complete once ASYNCcnt <= 16 (in-order).
            __builtin_amdgcn_s_wait_asynccnt(16);
#endif
        } else {
#if USE_ASYNC
            __builtin_amdgcn_s_wait_asynccnt(0);
#endif
        }
        __syncthreads();                    // current tile visible to all waves

        const __bf16* Kl = &Kt[cur][0];
        const __bf16* Vl = &Vt[cur][0];

        // ---- S^T = K Q^T : four 16x16 subtiles (j rows x i cols) ----
        // Double-buffered K fragments so ds_load latency hides under WMMA.
        v8f s[4];
#pragma unroll
        for (int jn = 0; jn < 4; ++jn)
#pragma unroll
            for (int e = 0; e < 8; ++e) s[jn][e] = 0.0f;

        v16bf kf[2][4];
#pragma unroll
        for (int jn = 0; jn < 4; ++jn)
            kf[0][jn] = load_frag(Kl + (jn * 16) * KLDE, KLDE);
#pragma unroll
        for (int kc = 0; kc < 4; ++kc) {
            const int cb = kc & 1;
            if (kc < 3) {
#pragma unroll
                for (int jn = 0; jn < 4; ++jn)
                    kf[cb ^ 1][jn] =
                        load_frag(Kl + (jn * 16) * KLDE + (kc + 1) * 32, KLDE);
            }
#pragma unroll
            for (int jn = 0; jn < 4; ++jn)
                s[jn] = __builtin_amdgcn_wmma_f32_16x16x32_bf16(
                    false, kf[cb][jn], false, qf[kc], (short)0, s[jn], false, false);
        }

        // Preload first V fragments; their ds latency hides under softmax VALU.
        v16bf vf[2][2];
        vf[0][0] = load_frag(Vl, VLDE);
        vf[0][1] = load_frag(Vl + 32, VLDE);

        // ---- lane-local online softmax (base-2; Q pre-scaled by log2 e) ----
        float tmax = -3.0e38f;
#pragma unroll
        for (int jn = 0; jn < 4; ++jn)
#pragma unroll
            for (int e = 0; e < 8; ++e) tmax = fmaxf(tmax, s[jn][e]);
        tmax = fmaxf(tmax, __shfl_xor(tmax, 16, 32));
        const float nm = fmaxf(mlane, tmax);
        const float alpha = __builtin_amdgcn_exp2f(mlane - nm);
        mlane = nm;

        float ps = 0.0f;
        v16bf pf0, pf1;
#pragma unroll
        for (int jn = 0; jn < 4; ++jn) {
#pragma unroll
            for (int e = 0; e < 8; ++e) {
                const float p = __builtin_amdgcn_exp2f(s[jn][e] - nm);
                ps += p;
                const __bf16 pb = (__bf16)p;
                // S^T accumulator is exactly the A-layout P fragment:
                // halves 0-7 <- subtile jn=0/2, halves 8-15 <- subtile jn=1/3.
                if (jn == 0)      pf0[e]     = pb;
                else if (jn == 1) pf0[8 + e] = pb;
                else if (jn == 2) pf1[e]     = pb;
                else              pf1[8 + e] = pb;
            }
        }
        llane = llane * alpha + ps;

        // ---- rescale O: broadcast alpha from stat lanes to O-layout rows ----
#pragma unroll
        for (int r = 0; r < 8; ++r) {
            const float ar = __shfl(alpha, r + rhalf, 16);
#pragma unroll
            for (int ct = 0; ct < 8; ++ct) O[ct][r] *= ar;
        }

        // ---- O += P V (double-buffered V fragments) ----
#pragma unroll
        for (int ct = 0; ct < 8; ++ct) {
            const int cb = ct & 1;
            if (ct < 7) {
                vf[cb ^ 1][0] = load_frag(Vl + (ct + 1) * 16 * VLDE, VLDE);
                vf[cb ^ 1][1] = load_frag(Vl + (ct + 1) * 16 * VLDE + 32, VLDE);
            }
            O[ct] = __builtin_amdgcn_wmma_f32_16x16x32_bf16(
                false, pf0, false, vf[cb][0], (short)0, O[ct], false, false);
            O[ct] = __builtin_amdgcn_wmma_f32_16x16x32_bf16(
                false, pf1, false, vf[cb][1], (short)0, O[ct], false, false);
        }
        __syncthreads();                    // protect cur buffers from next stage
    }

    // ---- epilogue: combine partner-lane partial sums, out = x + O/l ----
    // Per (lane, ct) the 8 r-values are contiguous in i: use b128 accesses.
    llane += __shfl_xor(llane, 16, 32);
    const float invl = 1.0f / llane;
    float ir[8];
#pragma unroll
    for (int r = 0; r < 8; ++r) ir[r] = __shfl(invl, r + rhalf, 16);

#pragma unroll
    for (int ct = 0; ct < 8; ++ct) {
        const int c = ct * 16 + colc;
        const size_t base = ((size_t)b * CC + c) * NN + i0 + rhalf;
        const float4 x0 = *reinterpret_cast<const float4*>(x + base);
        const float4 x1 = *reinterpret_cast<const float4*>(x + base + 4);
        float4 o0, o1;
        o0.x = x0.x + O[ct][0] * ir[0];
        o0.y = x0.y + O[ct][1] * ir[1];
        o0.z = x0.z + O[ct][2] * ir[2];
        o0.w = x0.w + O[ct][3] * ir[3];
        o1.x = x1.x + O[ct][4] * ir[4];
        o1.y = x1.y + O[ct][5] * ir[5];
        o1.z = x1.z + O[ct][6] * ir[6];
        o1.w = x1.w + O[ct][7] * ir[7];
        *reinterpret_cast<float4*>(out + base)     = o0;
        *reinterpret_cast<float4*>(out + base + 4) = o1;
    }
}

extern "C" void kernel_launch(void* const* d_in, const int* in_sizes, int n_in,
                              void* d_out, int out_size, void* d_ws, size_t ws_size,
                              hipStream_t stream)
{
    const float* x  = (const float*)d_in[0];
    const float* Wq = (const float*)d_in[1];
    const float* bq = (const float*)d_in[2];
    const float* Wk = (const float*)d_in[3];
    const float* bk = (const float*)d_in[4];
    const float* Wv = (const float*)d_in[5];
    const float* bv = (const float*)d_in[6];
    float* out = (float*)d_out;

    // Workspace: Q [B,N,C] + K [B,N,C] + V [B,C,N], bf16 (12 MB total).
    __bf16* qm = (__bf16*)d_ws;
    __bf16* km = qm + (size_t)BB * NN * CC;
    __bf16* vm = km + (size_t)BB * NN * CC;

    qkv_proj<<<BB * NN / 16, 128, 0, stream>>>(x, Wq, bq, Wk, bk, Wv, bv,
                                               qm, km, vm);
    attn_fa<<<(BB * NN / 16) / 4, 128, 0, stream>>>(qm, km, vm, x, out);
}
